// Network_67388036874689
// MI455X (gfx1250) — compile-verified, matrix operand order checked by reading, not run on hardware
//
#include <hip/hip_runtime.h>
#include <hip/hip_bf16.h>
#include <stdint.h>

// Problem constants (from reference): B=256, L=256, V=14, EMB=512, pooled H=128.
#define BATCH 256
#define LSEQ  256
#define VOCAB 14
#define HPOOL 128
#define HPAD  130   // pooled length + 1 zero pad row on each side (h-major layout)
#define EMB_C 512   // embedding width / enemy-friend conv input channels

typedef __attribute__((ext_vector_type(16))) __bf16 v16bf;
typedef __attribute__((ext_vector_type(8)))  float  v8f;
typedef __attribute__((ext_vector_type(4)))  unsigned int u32x4;
typedef __attribute__((ext_vector_type(8)))  int i32x8;
typedef __attribute__((ext_vector_type(4)))  int i32x4;

struct __align__(16) Chunk16 { uint32_t w[4]; };   // one 16-byte (8 x bf16) chunk
union ABFrag { Chunk16 c[2]; v16bf v; };           // 16x32 / 32x16 bf16 fragment (8 VGPRs)

__device__ __forceinline__ uint16_t f2bf(float f) {
  union { float f; uint32_t u; } cv; cv.f = f;
  uint32_t u = cv.u;
  u += 0x7FFFu + ((u >> 16) & 1u);                 // round-to-nearest-even
  return (uint16_t)(u >> 16);
}

// ---------------------------------------------------------------------------
// Tensor Data Mover: DMA a 2D bf16 tile [rows x kElems] (row stride = strideK
// elements) from global into LDS offset 0.  D# layout per CDNA5 ISA
// CDNA5 ISA.  Tracked by TENSORcnt.
// ---------------------------------------------------------------------------
__device__ __forceinline__ void tdm_load_panel(const uint16_t* gptr,
                                               int kElems, int rows, int strideK) {
  uint64_t ga = (uint64_t)(uintptr_t)gptr;
  u32x4 g0;
  g0[0] = 1u;                                            // count=1 valid user D#
  g0[1] = 0u;                                            // lds_addr = 0 (bytes)
  g0[2] = (uint32_t)ga;                                  // global_addr[31:0]
  g0[3] = (uint32_t)((ga >> 32) & 0x01FFFFFFu)           // global_addr[56:32]
        | (2u << 30);                                    // type = 2 ("image")
  uint32_t td0 = (uint32_t)strideK;                      // tensor_dim0
  uint32_t td1 = (uint32_t)rows;                         // tensor_dim1
  uint64_t st0 = (uint64_t)strideK;                      // tensor_dim0_stride
  i32x8 g1;
  g1[0] = (int)(1u << 16);                               // wg_mask=0, data_size=1 (2B)
  g1[1] = (int)((td0 & 0xFFFFu) << 16);                  // tensor_dim0[15:0]
  g1[2] = (int)(((td0 >> 16) & 0xFFFFu) | ((td1 & 0xFFFFu) << 16));
  g1[3] = (int)(((td1 >> 16) & 0xFFFFu) | (((uint32_t)kElems & 0xFFFFu) << 16)); // tile_dim0
  g1[4] = (int)((uint32_t)rows & 0xFFFFu);               // tile_dim1, tile_dim2=0
  g1[5] = (int)(uint32_t)(st0 & 0xFFFFFFFFu);            // dim0 stride lo
  g1[6] = (int)(uint32_t)((st0 >> 32) & 0xFFFFu);        // dim0 stride hi, dim1_stride=0
  g1[7] = 0;
  i32x4 z4 = {0, 0, 0, 0};
#if defined(__clang_major__) && (__clang_major__ >= 23)
  i32x8 z8 = {0, 0, 0, 0, 0, 0, 0, 0};
  __builtin_amdgcn_tensor_load_to_lds(g0, g1, z4, z4, z8, 0);
#else
  __builtin_amdgcn_tensor_load_to_lds(g0, g1, z4, z4, 0);
#endif
}

// ---------------------------------------------------------------------------
// Embedding gather + maxpool(k=2) -> bf16, h-major with zero pad rows.
// ---------------------------------------------------------------------------
__global__ void embed_pool_kernel(const int* __restrict__ tok,
                                  const float* __restrict__ emb,
                                  uint16_t* __restrict__ P) {
  int row = blockIdx.x;
  int b = row / HPAD, hp = row % HPAD;
  uint16_t* dst = P + (size_t)row * EMB_C;
  if (hp == 0 || hp == HPAD - 1) {
    for (int c = threadIdx.x; c < EMB_C; c += blockDim.x) dst[c] = 0;
    return;
  }
  int h  = hp - 1;
  int t0 = tok[b * LSEQ + 2 * h];
  int t1 = tok[b * LSEQ + 2 * h + 1];
  const float* e0 = emb + (size_t)t0 * EMB_C;
  const float* e1 = emb + (size_t)t1 * EMB_C;
  for (int c = threadIdx.x; c < EMB_C; c += blockDim.x)
    dst[c] = f2bf(fmaxf(e0[c], e1[c]));
}

// Conv weight repack: src [O,C,3,3] f32 -> dst [O, 3*C] bf16 taking kx==1 only.
__global__ void reorder_convw_kernel(const float* __restrict__ w,
                                     uint16_t* __restrict__ w2, int O, int C) {
  int i = blockIdx.x * blockDim.x + threadIdx.x;
  if (i >= O * 3 * C) return;
  int o = i / (3 * C);
  int r = i % (3 * C);
  int t = r / C, c = r % C;
  w2[i] = f2bf(w[((size_t)(o * C + c) * 3 + t) * 3 + 1]);
}

// f32 [K,N] -> bf16 transposed [N,K] (so GEMM B-tiles are contiguous K-runs).
__global__ void transpose_bf16_kernel(const float* __restrict__ src,
                                      uint16_t* __restrict__ dst, int K, int N) {
  int i = blockIdx.x * blockDim.x + threadIdx.x;
  if (i >= K * N) return;
  int n = i / K, k = i % K;
  dst[(size_t)n * K + k] = f2bf(src[(size_t)k * N + n]);
}

// ---------------------------------------------------------------------------
// 3-tap conv as WMMA GEMM.  One block = one (batch b, 32-row m-tile); 4 waves
// cover all 128 h positions.  The shared 32xK weight panel is DMA'd into LDS
// once via the Tensor Data Mover, A fragments then come from LDS; B columns
// are contiguous K-runs of the h-major padded activation blob.
// ---------------------------------------------------------------------------
__global__ void __launch_bounds__(128)
conv_gemm_kernel(const uint16_t* __restrict__ W2,
                 const uint16_t* __restrict__ P,
                 const float* __restrict__ bias,
                 uint16_t* __restrict__ out,
                 int M, int C, int relu) {
  extern __shared__ uint16_t ldsA[];   // 32 rows x K bf16 (96 KB when K=1536)
  const int K  = 3 * C;
  const int tmBlocks = M / 32;
  int b  = blockIdx.x / tmBlocks;
  int m0 = (blockIdx.x % tmBlocks) * 32;
  int waveId = threadIdx.x >> 5;
  int lane   = threadIdx.x & 31;
  int n0 = waveId * 32;

  if (waveId == 0) {
    tdm_load_panel(W2 + (size_t)m0 * K, K, 32, K);
    __builtin_amdgcn_s_wait_tensorcnt(0);
  }
  __syncthreads();

  int row = lane & 15;
  int hi  = lane >> 4;

  const uint16_t* Pb = P + (size_t)b * HPAD * C;
  v8f acc[2][2] = {};

  for (int k0 = 0; k0 < K; k0 += 32) {
    ABFrag a0, a1, b0, b1;
    const Chunk16* La0 = (const Chunk16*)(ldsA + (size_t)row * K + k0);
    const Chunk16* La1 = (const Chunk16*)(ldsA + (size_t)(16 + row) * K + k0);
    const Chunk16* Bp0 = (const Chunk16*)(Pb + (size_t)(n0 + row) * C + k0);
    const Chunk16* Bp1 = (const Chunk16*)(Pb + (size_t)(n0 + 16 + row) * C + k0);
    __builtin_prefetch((const void*)(Bp0 + 8), 0, 1);   // 2 K-steps ahead
    __builtin_prefetch((const void*)(Bp1 + 8), 0, 1);
    a0.c[0] = La0[hi];  a0.c[1] = La0[2 + hi];
    a1.c[0] = La1[hi];  a1.c[1] = La1[2 + hi];
    b0.c[0] = Bp0[hi];  b0.c[1] = Bp0[2 + hi];
    b1.c[0] = Bp1[hi];  b1.c[1] = Bp1[2 + hi];
    acc[0][0] = __builtin_amdgcn_wmma_f32_16x16x32_bf16(false, a0.v, false, b0.v, (short)0, acc[0][0], false, false);
    acc[0][1] = __builtin_amdgcn_wmma_f32_16x16x32_bf16(false, a0.v, false, b1.v, (short)0, acc[0][1], false, false);
    acc[1][0] = __builtin_amdgcn_wmma_f32_16x16x32_bf16(false, a1.v, false, b0.v, (short)0, acc[1][0], false, false);
    acc[1][1] = __builtin_amdgcn_wmma_f32_16x16x32_bf16(false, a1.v, false, b1.v, (short)0, acc[1][1], false, false);
  }

  int nlane = lane & 15;
  size_t ob = (size_t)b * ((size_t)M * HPOOL);
#pragma unroll
  for (int im = 0; im < 2; ++im)
#pragma unroll
    for (int in = 0; in < 2; ++in) {
      int mb = m0 + im * 16 + hi * 8;
      int nb = n0 + in * 16 + nlane;
#pragma unroll
      for (int rr = 0; rr < 8; ++rr) {
        int o = mb + rr;
        float v = acc[im][in][rr] + bias[o];
        if (relu) v = fmaxf(v, 0.f);
        out[ob + (size_t)o * HPOOL + nb] = f2bf(v);
      }
    }
}

// ---------------------------------------------------------------------------
// Split-K WMMA GEMM: Cpart[s][M,N] = A[M, slice s] * Bt[N, slice s]^T.
// ---------------------------------------------------------------------------
__global__ void __launch_bounds__(128)
gemm_splitk_kernel(const uint16_t* __restrict__ A,
                   const uint16_t* __restrict__ Bt,
                   float* __restrict__ Cpart,
                   int M, int N, int K, int Kslice) {
  const int tn = N / 32;
  const int tiles = (M / 32) * tn;
  int wave = blockIdx.x * (blockDim.x >> 5) + (threadIdx.x >> 5);
  int lane = threadIdx.x & 31;
  int s    = wave / tiles;
  int tile = wave % tiles;
  int m0 = (tile / tn) * 32;
  int n0 = (tile % tn) * 32;
  int kBeg = s * Kslice, kEnd = kBeg + Kslice;
  int row = lane & 15;
  int hi  = lane >> 4;

  v8f acc[2][2] = {};
  for (int k0 = kBeg; k0 < kEnd; k0 += 32) {
    ABFrag a0, a1, b0, b1;
    const Chunk16* Ap0 = (const Chunk16*)(A  + (size_t)(m0 + row) * K + k0);
    const Chunk16* Ap1 = (const Chunk16*)(A  + (size_t)(m0 + 16 + row) * K + k0);
    const Chunk16* Bp0 = (const Chunk16*)(Bt + (size_t)(n0 + row) * K + k0);
    const Chunk16* Bp1 = (const Chunk16*)(Bt + (size_t)(n0 + 16 + row) * K + k0);
    __builtin_prefetch((const void*)(Ap0 + 8), 0, 1);
    __builtin_prefetch((const void*)(Bp0 + 8), 0, 1);
    a0.c[0] = Ap0[hi];  a0.c[1] = Ap0[2 + hi];
    a1.c[0] = Ap1[hi];  a1.c[1] = Ap1[2 + hi];
    b0.c[0] = Bp0[hi];  b0.c[1] = Bp0[2 + hi];
    b1.c[0] = Bp1[hi];  b1.c[1] = Bp1[2 + hi];
    acc[0][0] = __builtin_amdgcn_wmma_f32_16x16x32_bf16(false, a0.v, false, b0.v, (short)0, acc[0][0], false, false);
    acc[0][1] = __builtin_amdgcn_wmma_f32_16x16x32_bf16(false, a0.v, false, b1.v, (short)0, acc[0][1], false, false);
    acc[1][0] = __builtin_amdgcn_wmma_f32_16x16x32_bf16(false, a1.v, false, b0.v, (short)0, acc[1][0], false, false);
    acc[1][1] = __builtin_amdgcn_wmma_f32_16x16x32_bf16(false, a1.v, false, b1.v, (short)0, acc[1][1], false, false);
  }

  int nlane = lane & 15;
  float* Cs = Cpart + (size_t)s * ((size_t)M * N);
#pragma unroll
  for (int im = 0; im < 2; ++im)
#pragma unroll
    for (int in = 0; in < 2; ++in) {
      int mb = m0 + im * 16 + hi * 8;
      int nb = n0 + in * 16 + nlane;
#pragma unroll
      for (int rr = 0; rr < 8; ++rr)
        Cs[(size_t)(mb + rr) * N + nb] = acc[im][in][rr];
    }
}

// Sum split-K partials + bias.
__global__ void reduce_bias_kernel(const float* __restrict__ part,
                                   const float* __restrict__ bias,
                                   float* __restrict__ Cout,
                                   int MN, int N, int S) {
  int i = blockIdx.x * blockDim.x + threadIdx.x;
  if (i >= MN) return;
  float acc = bias[i % N];
  for (int s = 0; s < S; ++s) acc += part[(size_t)s * MN + i];
  Cout[i] = acc;
}

// Row softmax over 128 entries.
__global__ void softmax128_kernel(const float* __restrict__ in,
                                  float* __restrict__ out) {
  __shared__ float red[128];
  int b = blockIdx.x, t = threadIdx.x;
  float v = in[b * 128 + t];
  red[t] = v; __syncthreads();
  for (int s = 64; s > 0; s >>= 1) { if (t < s) red[t] = fmaxf(red[t], red[t + s]); __syncthreads(); }
  float mx = red[0]; __syncthreads();
  float e = __expf(v - mx);
  red[t] = e; __syncthreads();
  for (int s = 64; s > 0; s >>= 1) { if (t < s) red[t] += red[t + s]; __syncthreads(); }
  out[b * 128 + t] = e / red[0];
}

// Broadcast enemy softmax row across H into padded h-major bf16 blob.
__global__ void broadcast_pm_kernel(const float* __restrict__ eo,
                                    uint16_t* __restrict__ Pm) {
  int rowIdx = blockIdx.x;
  int b = rowIdx / HPAD, hp = rowIdx % HPAD;
  int c = threadIdx.x;
  Pm[(size_t)rowIdx * 128 + c] =
      (hp == 0 || hp == HPAD - 1) ? (uint16_t)0 : f2bf(eo[b * 128 + c]);
}

__global__ void tokens_kernel(const float* __restrict__ m,
                              int* __restrict__ tok, int n) {
  int i = blockIdx.x * blockDim.x + threadIdx.x;
  if (i < n) tok[i] = ((int)floorf(fabsf(m[i]) * 100.0f)) % VOCAB;
}

// Tiny head: logits = f1 @ w2 + b2, row-softmax. One thread per row.
__global__ void final_kernel(const float* __restrict__ f1,
                             const float* __restrict__ w2,
                             const float* __restrict__ b2,
                             float* __restrict__ out) {
  int b = blockIdx.x * blockDim.x + threadIdx.x;
  if (b >= BATCH) return;
  float lg[VOCAB];
#pragma unroll
  for (int j = 0; j < VOCAB; ++j) lg[j] = b2[j];
  for (int k = 0; k < 128; ++k) {
    float a = f1[b * 128 + k];
#pragma unroll
    for (int j = 0; j < VOCAB; ++j) lg[j] += a * w2[k * VOCAB + j];
  }
  float mx = lg[0];
#pragma unroll
  for (int j = 1; j < VOCAB; ++j) mx = fmaxf(mx, lg[j]);
  float s = 0.f;
#pragma unroll
  for (int j = 0; j < VOCAB; ++j) { lg[j] = __expf(lg[j] - mx); s += lg[j]; }
#pragma unroll
  for (int j = 0; j < VOCAB; ++j) out[b * VOCAB + j] = lg[j] / s;
}

// ---------------------------------------------------------------------------
extern "C" void kernel_launch(void* const* d_in, const int* in_sizes, int n_in,
                              void* d_out, int out_size, void* d_ws, size_t ws_size,
                              hipStream_t stream) {
  (void)in_sizes; (void)n_in; (void)out_size; (void)ws_size;
  const int*   x     = (const int*)  d_in[0];
  const float* e_emb = (const float*)d_in[1];
  const float* e_cw  = (const float*)d_in[2];
  const float* e_cb  = (const float*)d_in[3];
  const float* e_lw  = (const float*)d_in[4];
  const float* e_lb  = (const float*)d_in[5];
  /* d_in[6] rand_proj is dead: fog_of_war's greedy loop returns arange(B). */
  const float* m_cw  = (const float*)d_in[7];
  const float* m_cb  = (const float*)d_in[8];
  const float* m_lw  = (const float*)d_in[9];
  const float* m_lb  = (const float*)d_in[10];
  const float* f_emb = (const float*)d_in[11];
  const float* f_cw  = (const float*)d_in[12];
  const float* f_cb  = (const float*)d_in[13];
  const float* f_l1w = (const float*)d_in[14];
  const float* f_l1b = (const float*)d_in[15];
  const float* f_l2w = (const float*)d_in[16];
  const float* f_l2b = (const float*)d_in[17];
  float* out = (float*)d_out;

  char* ws = (char*)d_ws;
  size_t off = 0;
  auto take = [&](size_t bytes) -> char* {
    char* p = ws + off; off += (bytes + 255) & ~(size_t)255; return p;
  };

  uint16_t* W2e   = (uint16_t*)take((size_t)256 * 1536 * 2);
  uint16_t* W2m   = (uint16_t*)take((size_t)64  * 384  * 2);
  uint16_t* W2f   = (uint16_t*)take((size_t)256 * 1536 * 2);
  uint16_t* LWeT  = (uint16_t*)take((size_t)128 * 32768 * 2);
  uint16_t* LWmT  = (uint16_t*)take((size_t)256 * 8192  * 2);
  uint16_t* LWfT  = (uint16_t*)take((size_t)128 * 32768 * 2);
  uint16_t* P512  = (uint16_t*)take((size_t)BATCH * HPAD * 512 * 2);
  uint16_t* Pm    = (uint16_t*)take((size_t)BATCH * HPAD * 128 * 2);
  uint16_t* convO = (uint16_t*)take((size_t)BATCH * 32768 * 2);
  float* Cpart    = (float*)take((size_t)16 * 256 * 128 * 4);   // split-K partials (shared)
  float* logitsE  = (float*)take((size_t)BATCH * 128 * 4);
  float* eoSoft   = (float*)take((size_t)BATCH * 128 * 4);
  float* mbuf     = (float*)take((size_t)BATCH * 256 * 4);
  int*   tokens   = (int*)  take((size_t)BATCH * 256 * 4);
  float* f1       = (float*)take((size_t)BATCH * 128 * 4);

  // --- weight prep (every call; deterministic) ---
  reorder_convw_kernel<<<(256*3*512 + 255)/256, 256, 0, stream>>>(e_cw, W2e, 256, 512);
  reorder_convw_kernel<<<(64 *3*128 + 255)/256, 256, 0, stream>>>(m_cw, W2m, 64, 128);
  reorder_convw_kernel<<<(256*3*512 + 255)/256, 256, 0, stream>>>(f_cw, W2f, 256, 512);
  transpose_bf16_kernel<<<(32768*128 + 255)/256, 256, 0, stream>>>(e_lw,  LWeT, 32768, 128);
  transpose_bf16_kernel<<<(8192 *256 + 255)/256, 256, 0, stream>>>(m_lw,  LWmT, 8192,  256);
  transpose_bf16_kernel<<<(32768*128 + 255)/256, 256, 0, stream>>>(f_l1w, LWfT, 32768, 128);

  // --- enemy branch ---
  embed_pool_kernel<<<BATCH * HPAD, 256, 0, stream>>>(x, e_emb, P512);
  conv_gemm_kernel<<<BATCH * 8, 128, 32 * 1536 * 2, stream>>>(W2e, P512, e_cb, convO, 256, 512, 0);
  gemm_splitk_kernel<<<128, 128, 0, stream>>>(convO, LWeT, Cpart, 256, 128, 32768, 2048);
  reduce_bias_kernel<<<(256*128 + 255)/256, 256, 0, stream>>>(Cpart, e_lb, logitsE, 256*128, 128, 16);
  softmax128_kernel<<<BATCH, 128, 0, stream>>>(logitsE, eoSoft);

  // --- manipulator ---
  broadcast_pm_kernel<<<BATCH * HPAD, 128, 0, stream>>>(eoSoft, Pm);
  conv_gemm_kernel<<<BATCH * 2, 128, 32 * 384 * 2, stream>>>(W2m, Pm, m_cb, convO, 64, 128, 1);
  gemm_splitk_kernel<<<64, 128, 0, stream>>>(convO, LWmT, Cpart, 256, 256, 8192, 2048);
  reduce_bias_kernel<<<(256*256 + 255)/256, 256, 0, stream>>>(Cpart, m_lb, mbuf, 256*256, 256, 4);
  tokens_kernel<<<(BATCH*256 + 255)/256, 256, 0, stream>>>(mbuf, tokens, BATCH * 256);

  // --- friend branch ---
  embed_pool_kernel<<<BATCH * HPAD, 256, 0, stream>>>(tokens, f_emb, P512);
  conv_gemm_kernel<<<BATCH * 8, 128, 32 * 1536 * 2, stream>>>(W2f, P512, f_cb, convO, 256, 512, 0);
  gemm_splitk_kernel<<<128, 128, 0, stream>>>(convO, LWfT, Cpart, 256, 128, 32768, 2048);
  reduce_bias_kernel<<<(256*128 + 255)/256, 256, 0, stream>>>(Cpart, f_l1b, f1, 256*128, 128, 16);
  final_kernel<<<1, 256, 0, stream>>>(f1, f_l2w, f_l2b, out);
}
